// GroupedQueryAttention_10754598109193
// MI455X (gfx1250) — matrix operand order repaired and checked by reading, not compiled, not run on hardware
//
#include <hip/hip_runtime.h>
#include <stdint.h>

#define DEV_INLINE __device__ __forceinline__

typedef __attribute__((ext_vector_type(16))) __bf16 bf16x16;
typedef __attribute__((ext_vector_type(8)))  float  f32x8;
typedef __attribute__((ext_vector_type(4)))  unsigned int u32x4;
typedef __attribute__((ext_vector_type(8)))  int   i32x8;
typedef __attribute__((ext_vector_type(4)))  int   i32x4;

constexpr int B_   = 2;
constexpr int T_   = 2048;
constexpr int C_   = 2048;   // EMBED_DIM
constexpr int H_   = 32;
constexpr int KVH_ = 8;
constexpr int D_   = 64;
constexpr int KVD_ = 512;    // KVH*D

DEV_INLINE uint16_t f2bf(float f) {
  uint32_t u = __float_as_uint(f);
  u += 0x7FFFu + ((u >> 16) & 1u);          // round-to-nearest-even
  return (uint16_t)(u >> 16);
}
DEV_INLINE float bf2f(uint16_t h) { return __uint_as_float(((uint32_t)h) << 16); }

union FragCast { bf16x16 v; uint4 q[2]; };

// A-operand (16 rows x 32 K) from row-major [row][k] (ISA 7.12.2 16-bit A layout):
// lanes 0-15: row=lane, K = {0..7, 16..23}; lanes 16-31: row=lane-16, K = {8..15, 24..31}
DEV_INLINE bf16x16 load_a_frag(const uint16_t* base, int ld) {
  const int lane = threadIdx.x & 31;
  const int r = lane & 15, hi = lane >> 4;
  const uint16_t* p = base + (size_t)r * ld + hi * 8;
  FragCast u;
  u.q[0] = *(const uint4*)(p);
  u.q[1] = *(const uint4*)(p + 16);
  return u.v;
}
// B-operand (32 K x 16 cols) from row-major [col][k]:
// lanes 0-15: col=lane, K = {0..7, 16..23}; lanes 16-31: col=lane-16, K = {8..15, 24..31}
DEV_INLINE bf16x16 load_b_frag(const uint16_t* base, int ld) {
  const int lane = threadIdx.x & 31;
  const int r = lane & 15, hi = lane >> 4;
  const uint16_t* p = base + (size_t)r * ld + hi * 8;
  FragCast u;
  u.q[0] = *(const uint4*)(p);
  u.q[1] = *(const uint4*)(p + 16);
  return u.v;
}

DEV_INLINE f32x8 wmma_bf16(bf16x16 a, bf16x16 b, f32x8 c) {
  return __builtin_amdgcn_wmma_f32_16x16x32_bf16(false, a, false, b, (short)0, c,
                                                 false, false);
}

// ---------------------------------------------------------------- converts ---

__global__ void convert_f32_bf16_kernel(const float* __restrict__ in,
                                        uint16_t* __restrict__ out, int n) {
  int i = blockIdx.x * blockDim.x + threadIdx.x;
  if (i < n) out[i] = f2bf(in[i]);
}

// in: fp32 [R][Cc]  ->  out: bf16 [Cc][R]  (transpose + convert)
__global__ __launch_bounds__(256) void transpose_f32_bf16_kernel(
    const float* __restrict__ in, uint16_t* __restrict__ out, int R, int Cc) {
  __shared__ float tile[32][33];
  const int bx = blockIdx.x * 32, by = blockIdx.y * 32;
  const int tx = threadIdx.x, ty = threadIdx.y;
#pragma unroll
  for (int k = 0; k < 4; ++k)
    tile[ty + k * 8][tx] = in[(size_t)(by + ty + k * 8) * Cc + bx + tx];
  __syncthreads();
#pragma unroll
  for (int k = 0; k < 4; ++k)
    out[(size_t)(bx + ty + k * 8) * R + by + tx] = f2bf(tile[tx][ty + k * 8]);
}

// ------------------------------------------------------------------- GEMM ----
// C[M][N] = A[M][K] * Bt[N][K]^T ; bf16 in, f32 accumulate, bf16 or f32 out.
// Block = 128x128, 8 waves (4x2), each wave 32x64 (2x4 WMMA tiles).
// K staged 32 at a time, double-buffered LDS, Tensor Data Mover staging.

#define LDSS 40  // LDS row stride (elements): 20 dwords -> conflict-free frag reads
#define BM 128
#define BN 128

#if __has_builtin(__builtin_amdgcn_tensor_load_to_lds)
#define HAVE_TDM 1
#else
#define HAVE_TDM 0
#endif

#if HAVE_TDM
// One 2D-tile DMA descriptor: tile = rows x 32 bf16 elems, row stride `ld` elems,
// LDS rows padded 64B->80B (pad_interval=16 dwords, pad_amount=4 dwords) == LDSS.
DEV_INLINE void tdm_load_tile(const uint16_t* gbase, int ld, int rows,
                              uint32_t lds_off) {
  const uint64_t ga = (uint64_t)(uintptr_t)gbase;
  u32x4 g0;
  g0.x = 1u;                                   // count=1, user mode, no gather
  g0.y = lds_off;                              // lds_addr
  g0.z = (uint32_t)ga;                         // global_addr[31:0]
  g0.w = (uint32_t)(ga >> 32) | (2u << 30);    // global_addr[56:32] | type=2
  i32x8 g1;
  g1[0] = (1 << 16)            // data_size = 2 bytes
        | (1 << 20)            // pad_enable
        | (3 << 22)            // pad_interval: 16 dwords (64B row)
        | (3 << 25);           // pad_amount:   4 dwords (16B pad)
  g1[1] = (32 & 0xFFFF) << 16;               // tensor_dim0 = 32 (lo 16)
  g1[2] = ((rows & 0xFFFF) << 16);           // dim0 hi=0 | tensor_dim1 lo
  g1[3] = ((rows >> 16) & 0xFFFF) | (32 << 16); // tensor_dim1 hi | tile_dim0=32
  g1[4] = rows & 0xFFFF;                     // tile_dim1 = rows, tile_dim2=0
  g1[5] = ld;                                // tensor_dim0_stride lo (elements)
  g1[6] = 0;                                 // stride hi | dim1_stride lo
  g1[7] = 0;
  // clang-23 / therock form: 6 args (g0, g1, g2, g3, g4, cpol)
  __builtin_amdgcn_tensor_load_to_lds(g0, g1, (i32x4)0, (i32x4)0, (i32x8)0, 0);
}
#endif

__global__ __launch_bounds__(256) void gemm_bf16_kernel(
    const uint16_t* __restrict__ A, const uint16_t* __restrict__ Bt,
    void* __restrict__ Cout, int M, int N, int K,
    int lda, int ldb, int ldc, int out_f32) {
  __shared__ __align__(16) uint16_t As[2][BM * LDSS];
  __shared__ __align__(16) uint16_t Bs[2][BN * LDSS];
  const int n0 = blockIdx.x * BN;
  const int m0 = blockIdx.y * BM;
  const int tid = threadIdx.x;
  const int wave = tid >> 5;
  const int wm = wave >> 1;          // 0..3 -> 32-row strip
  const int wn = wave & 1;           // 0..1 -> 64-col strip
  const int lane = tid & 31;
  const int lr = lane & 15, hi = lane >> 4;

  f32x8 acc[2][4] = {};

  const uint16_t* Ag = A + (size_t)m0 * lda;
  const uint16_t* Bg = Bt + (size_t)n0 * ldb;

#if HAVE_TDM
  const uint32_t ldsA[2] = {(uint32_t)(uintptr_t)&As[0][0],
                            (uint32_t)(uintptr_t)&As[1][0]};
  const uint32_t ldsB[2] = {(uint32_t)(uintptr_t)&Bs[0][0],
                            (uint32_t)(uintptr_t)&Bs[1][0]};
  if (wave == 0) {                       // prologue: stage tile 0
    tdm_load_tile(Ag, lda, BM, ldsA[0]);
    tdm_load_tile(Bg, ldb, BN, ldsB[0]);
  }
#else
  {                                      // prologue: register-staged tile 0
#pragma unroll
    for (int t = 0; t < 2; ++t) {
      int v = tid + t * 256;             // 512 b128 slots per matrix
      int row = v >> 2, cv = (v & 3) * 8;
      *(uint4*)&As[0][row * LDSS + cv] = *(const uint4*)(Ag + (size_t)row * lda + cv);
      *(uint4*)&Bs[0][row * LDSS + cv] = *(const uint4*)(Bg + (size_t)row * ldb + cv);
    }
  }
#endif

  int buf = 0;
  for (int k0 = 0; k0 < K; k0 += 32) {
    const bool has_next = (k0 + 32) < K;
#if HAVE_TDM
    if (wave == 0) {
      if (has_next) {                    // stage next tile into other buffer
        tdm_load_tile(Ag + k0 + 32, lda, BM, ldsA[buf ^ 1]);
        tdm_load_tile(Bg + k0 + 32, ldb, BN, ldsB[buf ^ 1]);
        __builtin_amdgcn_s_wait_tensorcnt(2);   // current tile (oldest 2) done
      } else {
        __builtin_amdgcn_s_wait_tensorcnt(0);
      }
    }
#else
    if (has_next) {
      uint4 ra[2], rb[2];
#pragma unroll
      for (int t = 0; t < 2; ++t) {
        int v = tid + t * 256;
        int row = v >> 2, cv = (v & 3) * 8;
        ra[t] = *(const uint4*)(Ag + (size_t)row * lda + k0 + 32 + cv);
        rb[t] = *(const uint4*)(Bg + (size_t)row * ldb + k0 + 32 + cv);
      }
#pragma unroll
      for (int t = 0; t < 2; ++t) {
        int v = tid + t * 256;
        int row = v >> 2, cv = (v & 3) * 8;
        *(uint4*)&As[buf ^ 1][row * LDSS + cv] = ra[t];
        *(uint4*)&Bs[buf ^ 1][row * LDSS + cv] = rb[t];
      }
    }
#endif
    __syncthreads();

    const uint16_t* pa = &As[buf][(wm * 32) * LDSS];
    bf16x16 a0 = load_a_frag(pa, LDSS);
    bf16x16 a1 = load_a_frag(pa + 16 * LDSS, LDSS);
    bf16x16 b0 = load_b_frag(&Bs[buf][(wn * 64 + 0)  * LDSS], LDSS);
    bf16x16 b1 = load_b_frag(&Bs[buf][(wn * 64 + 16) * LDSS], LDSS);
    bf16x16 b2 = load_b_frag(&Bs[buf][(wn * 64 + 32) * LDSS], LDSS);
    bf16x16 b3 = load_b_frag(&Bs[buf][(wn * 64 + 48) * LDSS], LDSS);
    acc[0][0] = wmma_bf16(a0, b0, acc[0][0]);
    acc[0][1] = wmma_bf16(a0, b1, acc[0][1]);
    acc[0][2] = wmma_bf16(a0, b2, acc[0][2]);
    acc[0][3] = wmma_bf16(a0, b3, acc[0][3]);
    acc[1][0] = wmma_bf16(a1, b0, acc[1][0]);
    acc[1][1] = wmma_bf16(a1, b1, acc[1][1]);
    acc[1][2] = wmma_bf16(a1, b2, acc[1][2]);
    acc[1][3] = wmma_bf16(a1, b3, acc[1][3]);
    __syncthreads();
    buf ^= 1;
  }

#pragma unroll
  for (int i = 0; i < 2; ++i)
#pragma unroll
    for (int j = 0; j < 4; ++j) {
      const int rowb = m0 + wm * 32 + i * 16 + hi * 8;   // C-frag: row = r + hi*8
      const int col  = n0 + wn * 64 + j * 16 + lr;       //         col = lane&15
      if (out_f32) {
        float* C = (float*)Cout;
#pragma unroll
        for (int r = 0; r < 8; ++r) C[(size_t)(rowb + r) * ldc + col] = acc[i][j][r];
      } else {
        uint16_t* C = (uint16_t*)Cout;
#pragma unroll
        for (int r = 0; r < 8; ++r) C[(size_t)(rowb + r) * ldc + col] = f2bf(acc[i][j][r]);
      }
    }
}

// ------------------------------------------------------------------- RoPE ----
// Q: bf16 [B*T][C] (per-head cols), K: bf16 [B*T][KVD]. Q pre-scaled by D^-0.5.

__global__ void rope_kernel(uint16_t* __restrict__ Q, uint16_t* __restrict__ Km) {
  const int NQ = B_ * T_ * H_ * (D_ / 2);
  const int NK = B_ * T_ * KVH_ * (D_ / 2);
  int idx = blockIdx.x * blockDim.x + threadIdx.x;
  uint16_t* base;
  int i, t;
  float scl;
  size_t addr;
  if (idx < NQ) {
    i = idx & 31;
    int h = (idx >> 5) & (H_ - 1);
    t = (idx >> 10) & (T_ - 1);
    int b = idx >> 21;
    addr = (size_t)(b * T_ + t) * C_ + h * D_ + i;
    base = Q;
    scl = 0.125f;                           // D^-0.5, folded into Q
  } else {
    int j = idx - NQ;
    if (j >= NK) return;
    i = j & 31;
    int h = (j >> 5) & (KVH_ - 1);
    t = (j >> 8) & (T_ - 1);
    int b = j >> 19;
    addr = (size_t)(b * T_ + t) * KVD_ + h * D_ + i;
    base = Km;
    scl = 1.0f;
  }
  float inv = __expf(-(float)i * (9.2103403720f / 32.0f));   // 10000^(-i/32)
  float th = (float)t * inv;
  float sth, cth;
  __sincosf(th, &sth, &cth);
  float v0 = bf2f(base[addr]);
  float v1 = bf2f(base[addr + 32]);
  base[addr]      = f2bf((v0 * cth - v1 * sth) * scl);
  base[addr + 32] = f2bf((v1 * cth + v0 * sth) * scl);
}

// -------------------------------------------------------- flash attention ----
// One wave per 16-row Q tile; streams 32 keys/iter; full (non-causal) softmax.
// Q: [B*T][C], K: [B*T][KVD], Vt: [B][KVD][T], Y: [B*T][C] (all bf16)

__global__ __launch_bounds__(128) void attn_kernel(
    const uint16_t* __restrict__ Q, const uint16_t* __restrict__ Km,
    const uint16_t* __restrict__ Vt, uint16_t* __restrict__ Y) {
  __shared__ __align__(16) uint16_t P[4][16 * LDSS];
  const int tid = threadIdx.x;
  const int wave = tid >> 5;
  const int lane = tid & 31;
  const int lr = lane & 15, hi = lane >> 4;

  const int tile = blockIdx.x * 4 + wave;   // B*H*(T/16) tiles total
  const int qt = tile & 127;                // T/16 = 128
  const int h  = (tile >> 7) & (H_ - 1);
  const int b  = tile >> 12;
  const int kvh = h >> 2;                   // rep = H/KVH = 4

  const uint16_t* Qp = Q + (size_t)(b * T_ + qt * 16) * C_ + h * D_;
  const uint16_t* Kp = Km + (size_t)(b * T_) * KVD_ + kvh * D_;
  const uint16_t* Vp = Vt + (size_t)b * KVD_ * T_ + (size_t)(kvh * D_) * T_;
  uint16_t* Yp = Y + (size_t)(b * T_ + qt * 16) * C_ + h * D_;

  const bf16x16 qa0 = load_a_frag(Qp, C_);        // d = 0..31
  const bf16x16 qa1 = load_a_frag(Qp + 32, C_);   // d = 32..63

  f32x8 o[4] = {};
  float m[8], l[8];
#pragma unroll
  for (int r = 0; r < 8; ++r) { m[r] = -3.0e38f; l[r] = 0.f; }

  uint16_t* Pw = P[wave];

  for (int s = 0; s < T_; s += 32) {
    f32x8 s0 = {}, s1 = {};
    bf16x16 k00 = load_b_frag(Kp + (size_t)s * KVD_, KVD_);
    bf16x16 k01 = load_b_frag(Kp + (size_t)s * KVD_ + 32, KVD_);
    bf16x16 k10 = load_b_frag(Kp + (size_t)(s + 16) * KVD_, KVD_);
    bf16x16 k11 = load_b_frag(Kp + (size_t)(s + 16) * KVD_ + 32, KVD_);
    if (s + 32 < T_) {                       // WGP-local prefetch of next block
      __builtin_prefetch(Kp + (size_t)(s + 32 + lane) * KVD_, 0, 3);
      __builtin_prefetch(Vp + (size_t)lane * T_ + s + 32, 0, 3);
      __builtin_prefetch(Vp + (size_t)(lane + 32) * T_ + s + 32, 0, 3);
    }
    s0 = wmma_bf16(qa0, k00, s0);
    s0 = wmma_bf16(qa1, k01, s0);   // keys s..s+15
    s1 = wmma_bf16(qa0, k10, s1);
    s1 = wmma_bf16(qa1, k11, s1);   // keys s+16..s+31

    // online softmax: per C-frag row (r + hi*8), cols across 16 lanes
#pragma unroll
    for (int r = 0; r < 8; ++r) {
      float mx = fmaxf(s0[r], s1[r]);
#pragma unroll
      for (int off = 1; off < 16; off <<= 1) mx = fmaxf(mx, __shfl_xor(mx, off, 32));
      float mn = fmaxf(m[r], mx);
      float alpha = __expf(m[r] - mn);
      float p0 = __expf(s0[r] - mn);
      float p1 = __expf(s1[r] - mn);
      float rs = p0 + p1;
#pragma unroll
      for (int off = 1; off < 16; off <<= 1) rs += __shfl_xor(rs, off, 32);
      l[r] = l[r] * alpha + rs;
      m[r] = mn;
#pragma unroll
      for (int j = 0; j < 4; ++j) o[j][r] *= alpha;
      const int row = r + hi * 8;
      Pw[row * LDSS + lr]      = f2bf(p0);
      Pw[row * LDSS + 16 + lr] = f2bf(p1);
    }
    asm volatile("s_wait_dscnt 0" ::: "memory");  // P staged -> A-operand reload
    bf16x16 pa = load_a_frag(Pw, LDSS);           // 16 q-rows x 32 keys
#pragma unroll
    for (int j = 0; j < 4; ++j) {                 // O[:, j*16..j*16+15] += P*V
      bf16x16 vf = load_b_frag(Vp + (size_t)(j * 16) * T_ + s, T_);
      o[j] = wmma_bf16(pa, vf, o[j]);
    }
  }

#pragma unroll
  for (int r = 0; r < 8; ++r) {
    const float invl = 1.f / l[r];
    const int row = r + hi * 8;
#pragma unroll
    for (int j = 0; j < 4; ++j)
      Yp[(size_t)row * C_ + j * 16 + lr] = f2bf(o[j][r] * invl);
  }
}

// ---------------------------------------------------------------- launch -----

extern "C" void kernel_launch(void* const* d_in, const int* in_sizes, int n_in,
                              void* d_out, int out_size, void* d_ws, size_t ws_size,
                              hipStream_t stream) {
  const float* x  = (const float*)d_in[0];
  const float* Wq = (const float*)d_in[1];
  const float* Wk = (const float*)d_in[2];
  const float* Wv = (const float*)d_in[3];
  const float* Wo = (const float*)d_in[4];

  char* w = (char*)d_ws;
  auto alloc = [&](size_t bytes) -> void* {
    void* p = (void*)w;
    w += (bytes + 255) & ~(size_t)255;
    return p;
  };
  uint16_t* xb  = (uint16_t*)alloc((size_t)B_ * T_ * C_ * 2);
  uint16_t* WqT = (uint16_t*)alloc((size_t)C_ * C_ * 2);
  uint16_t* WkT = (uint16_t*)alloc((size_t)KVD_ * C_ * 2);
  uint16_t* WvT = (uint16_t*)alloc((size_t)KVD_ * C_ * 2);
  uint16_t* WoT = (uint16_t*)alloc((size_t)C_ * C_ * 2);
  uint16_t* Qb  = (uint16_t*)alloc((size_t)B_ * T_ * C_ * 2);
  uint16_t* Kb  = (uint16_t*)alloc((size_t)B_ * T_ * KVD_ * 2);
  uint16_t* Vtb = (uint16_t*)alloc((size_t)B_ * KVD_ * T_ * 2);
  uint16_t* Yb  = (uint16_t*)alloc((size_t)B_ * T_ * C_ * 2);
  (void)ws_size; (void)n_in; (void)in_sizes; (void)out_size;

  {  // x -> bf16
    int n = B_ * T_ * C_;
    convert_f32_bf16_kernel<<<(n + 255) / 256, 256, 0, stream>>>(x, xb, n);
  }
  {  // weights -> transposed bf16 [N][K]
    dim3 blk(32, 8);
    transpose_f32_bf16_kernel<<<dim3(C_ / 32, C_ / 32), blk, 0, stream>>>(Wq, WqT, C_, C_);
    transpose_f32_bf16_kernel<<<dim3(KVD_ / 32, C_ / 32), blk, 0, stream>>>(Wk, WkT, C_, KVD_);
    transpose_f32_bf16_kernel<<<dim3(KVD_ / 32, C_ / 32), blk, 0, stream>>>(Wv, WvT, C_, KVD_);
    transpose_f32_bf16_kernel<<<dim3(C_ / 32, C_ / 32), blk, 0, stream>>>(Wo, WoT, C_, C_);
  }
  {  // Q = x @ Wq    -> bf16 [B*T][C]
    dim3 g(C_ / BN, (B_ * T_) / BM);
    gemm_bf16_kernel<<<g, 256, 0, stream>>>(xb, WqT, Qb, B_ * T_, C_, C_, C_, C_, C_, 0);
  }
  {  // K = x @ Wk    -> bf16 [B*T][KVD]
    dim3 g(KVD_ / BN, (B_ * T_) / BM);
    gemm_bf16_kernel<<<g, 256, 0, stream>>>(xb, WkT, Kb, B_ * T_, KVD_, C_, C_, C_, KVD_, 0);
  }
  for (int b = 0; b < B_; ++b) {  // V^T = Wv^T @ x^T  -> bf16 [KVD][T] per batch
    dim3 g(T_ / BN, KVD_ / BM);
    gemm_bf16_kernel<<<g, 256, 0, stream>>>(WvT, xb + (size_t)b * T_ * C_,
                                            Vtb + (size_t)b * KVD_ * T_,
                                            KVD_, T_, C_, C_, C_, T_, 0);
  }
  {  // RoPE on Q (pre-scaled) and K
    int n = B_ * T_ * H_ * (D_ / 2) + B_ * T_ * KVH_ * (D_ / 2);
    rope_kernel<<<(n + 255) / 256, 256, 0, stream>>>(Qb, Kb);
  }
  {  // flash attention -> Y bf16
    attn_kernel<<<(B_ * H_ * (T_ / 16)) / 4, 128, 0, stream>>>(Qb, Kb, Vtb, Yb);
  }
  {  // out = Y @ Wo  -> fp32 d_out
    dim3 g(C_ / BN, (B_ * T_) / BM);
    gemm_bf16_kernel<<<g, 256, 0, stream>>>(Yb, WoT, d_out, B_ * T_, C_, C_, C_, C_, C_, 1);
  }
}
